// LSTM_78280073937672
// MI455X (gfx1250) — matrix operand order repaired and checked by reading, not compiled, not run on hardware
//
#include <hip/hip_runtime.h>
#include <hip/hip_bf16.h>

typedef _Float16     half16_t __attribute__((ext_vector_type(16)));
typedef float        float8_t __attribute__((ext_vector_type(8)));
typedef unsigned int u32x4    __attribute__((ext_vector_type(4)));

union FragU { half16_t h; unsigned int u[8]; };

__device__ __forceinline__ float8_t wmma16(half16_t a, half16_t b, float8_t c) {
  // D = A(16x32 f16) * B(32x16 f16) + C(16x16 f32)
  return __builtin_amdgcn_wmma_f32_16x16x32_f16(false, a, false, b, (short)0, c, false, false);
}

__device__ __forceinline__ unsigned int pack2f16(float a, float b) {
  union { _Float16 h[2]; unsigned int u; } q;
  q.h[0] = (_Float16)a; q.h[1] = (_Float16)b;
  return q.u;
}

#if __has_builtin(__builtin_amdgcn_tanhf)
__device__ __forceinline__ float tanh1(float x) { return __builtin_amdgcn_tanhf(x); }
__device__ __forceinline__ float sig1(float x)  { return __builtin_amdgcn_tanhf(0.5f * x) * 0.5f + 0.5f; }
#else
__device__ __forceinline__ float tanh1(float x) { return 1.0f - 2.0f * __builtin_amdgcn_rcpf(__expf(2.0f * x) + 1.0f); }
__device__ __forceinline__ float sig1(float x)  { return __builtin_amdgcn_rcpf(1.0f + __expf(-x)); }
#endif

// 16x64 f16 matrix in LDS (row-major [M][k]) -> two WMMA-A fragments.
// A layout: lane L: M=L&15, half=L>>4; k-pairs are 16B-contiguous per half-wave,
// so each fragment half is one ds_load_b128.
__device__ __forceinline__ void loadA64(const _Float16* hs, int laneM, int hf,
                                        half16_t& lo, half16_t& hi) {
  const u32x4* hq = (const u32x4*)hs;      // 8 x 16B chunks per row
  int base = laneM * 8 + hf;
  u32x4 q0 = hq[base];                      // k-pairs p+4hf        -> u[0..3]
  u32x4 q1 = hq[base + 2];                  // k-pairs 8+p+4hf      -> u[4..7]
  u32x4 q2 = hq[base + 4];                  // same, k+32
  u32x4 q3 = hq[base + 6];
  FragU l, h;
  l.u[0] = q0.x; l.u[1] = q0.y; l.u[2] = q0.z; l.u[3] = q0.w;
  l.u[4] = q1.x; l.u[5] = q1.y; l.u[6] = q1.z; l.u[7] = q1.w;
  h.u[0] = q2.x; h.u[1] = q2.y; h.u[2] = q2.z; h.u[3] = q2.w;
  h.u[4] = q3.x; h.u[5] = q3.y; h.u[6] = q3.z; h.u[7] = q3.w;
  lo = l.h; hi = h.h;
}

// Weights in LDS pre-packed as u32 = (Wt[k][n], Wt[k+1][n]) at [kpair*256 + n].
// B layout: lane L: n=n0+(L&15), half=L>>4; VGPR p<4: k=2p+8*half, p>=4: k=16+2(p-4)+8*half.
__device__ __forceinline__ half16_t loadB32(const unsigned int* Wp, int kb, int n, int hf) {
  FragU b;
#pragma unroll
  for (int p = 0; p < 8; ++p)
    b.u[p] = Wp[(kb * 16 + p + (p & 4) + (hf << 2)) * 256 + n];
  return b.h;
}

// acc[g][r] holds gate g at (M = r + 8*half, hidden j).  Cell update in registers,
// new h written f16 to the write ping-pong buffer.
__device__ __forceinline__ void gates_to_h(const float8_t* acc, float* creg,
                                           _Float16* hs, int j, int hf) {
#pragma unroll
  for (int r = 0; r < 8; ++r) {
    float ig = sig1(acc[0][r]);
    float fg = sig1(acc[1][r]);
    float gg = tanh1(acc[2][r]);
    float og = sig1(acc[3][r]);
    float c  = fg * creg[r] + ig * gg;
    creg[r]  = c;
    hs[((r + (hf << 3)) << 6) + j] = (_Float16)(og * tanh1(c));
  }
}

__global__ __launch_bounds__(128, 1) void lstm_fused_kernel(
    const float* __restrict__ x,
    const float* __restrict__ Wih0, const float* __restrict__ Whh0,
    const float* __restrict__ bih0, const float* __restrict__ bhh0,
    const float* __restrict__ Wih1, const float* __restrict__ Whh1,
    const float* __restrict__ bih1, const float* __restrict__ bhh1,
    const float* __restrict__ W1,  const float* __restrict__ b1l,
    const float* __restrict__ gma, const float* __restrict__ bta,
    const float* __restrict__ rm,  const float* __restrict__ rv,
    const float* __restrict__ W2,  const float* __restrict__ b2,
    float* __restrict__ out) {
  // ---- LDS: packed weight staging + double-buffered state (~109 KB) ----
  __shared__ unsigned int Whh0p[32 * 256];
  __shared__ unsigned int Wih1p[32 * 256];
  __shared__ unsigned int Whh1p[32 * 256];
  __shared__ unsigned int Wih0p[4 * 256];
  __shared__ __align__(16) _Float16 h0s0[1024], h0s1[1024];
  __shared__ __align__(16) _Float16 h1s0[1024], h1s1[1024];
  __shared__ __align__(16) unsigned int xs0[64], xs1[64];

  const int tid   = threadIdx.x;
  const int lane  = tid & 31;
  const int wv    = tid >> 5;
  const int laneM = lane & 15;
  const int hf    = lane >> 4;
  const int j     = (wv << 4) + laneM;
  const int wg    = blockIdx.x;
  const long xbase = (long)(wg * 16) * (7 * 2048);

  // ---- one-time: pack weights into LDS, zero state, prime x pipeline ----
  for (int idx = tid; idx < 32 * 256; idx += 128) {
    int kp = idx >> 8, n = idx & 255, k = kp << 1;
    Whh0p[idx] = pack2f16(Whh0[n * 64 + k], Whh0[n * 64 + k + 1]);
    Wih1p[idx] = pack2f16(Wih1[n * 64 + k], Wih1[n * 64 + k + 1]);
    Whh1p[idx] = pack2f16(Whh1[n * 64 + k], Whh1[n * 64 + k + 1]);
  }
  for (int idx = tid; idx < 4 * 256; idx += 128) {
    int kp = idx >> 8, n = idx & 255, k = kp << 1;
    float a = (k < 7)     ? Wih0[n * 7 + k]     : 0.f;
    float b = (k + 1 < 7) ? Wih0[n * 7 + k + 1] : 0.f;
    Wih0p[idx] = pack2f16(a, b);
  }
  for (int idx = tid; idx < 1024; idx += 128) {
    h0s0[idx] = (_Float16)0.f; h0s1[idx] = (_Float16)0.f;
    h1s0[idx] = (_Float16)0.f; h1s1[idx] = (_Float16)0.f;
  }

  float rx0 = 0.f, rx1 = 0.f;       // x pipeline registers (data for step t+1)
  if (tid < 64) {
    int row = tid >> 2, pp = tid & 3, k0 = pp << 1;
    const float* xr = x + xbase + (long)row * (7 * 2048);
    float f0 = xr[k0 * 2048 + 0];
    float f1 = (k0 + 1 < 7) ? xr[(k0 + 1) * 2048 + 0] : 0.f;
    xs0[tid] = pack2f16(f0, f1);    // t = 0
    rx0 = xr[k0 * 2048 + 1];        // t = 1
    rx1 = (k0 + 1 < 7) ? xr[(k0 + 1) * 2048 + 1] : 0.f;
  }

  float c0r[8], c1r[8];
#pragma unroll
  for (int r = 0; r < 8; ++r) { c0r[r] = 0.f; c1r[r] = 0.f; }

  __syncthreads();

  // ---- hoist ALL weight B-fragments + bias splat vectors into registers ----
  half16_t bWx[4];                 // Wih0 (K padded to 32)
  half16_t bW0[4][2];              // Whh0
  half16_t bW1i[4][2];             // Wih1
  half16_t bW1h[4][2];             // Whh1
  float8_t b0v[4], b1v[4];
#pragma unroll
  for (int g = 0; g < 4; ++g) {
    int n = (g << 6) + j;
    FragU bx;
#pragma unroll
    for (int p = 0; p < 8; ++p)
      bx.u[p] = (hf == 0 && p < 4) ? Wih0p[(p << 8) + n] : 0u;
    bWx[g]     = bx.h;
    bW0[g][0]  = loadB32(Whh0p, 0, n, hf);
    bW0[g][1]  = loadB32(Whh0p, 1, n, hf);
    bW1i[g][0] = loadB32(Wih1p, 0, n, hf);
    bW1i[g][1] = loadB32(Wih1p, 1, n, hf);
    bW1h[g][0] = loadB32(Whh1p, 0, n, hf);
    bW1h[g][1] = loadB32(Whh1p, 1, n, hf);
    float bb0 = bih0[n] + bhh0[n];
    float bb1 = bih1[n] + bhh1[n];
#pragma unroll
    for (int r = 0; r < 8; ++r) { b0v[g][r] = bb0; b1v[g][r] = bb1; }
  }

  // One step: reads h*[rd], writes h*[wr]; stages x(t+1) from regs, prefetches x(t+2).
  // Single barrier per step: every cross-step RAW is separated by >=1 barrier;
  // ping-pong buffers remove all WAR hazards.
  auto step = [&](int t, const unsigned int* xs_rd, unsigned int* xs_wr,
                  const _Float16* h0_rd, _Float16* h0_wr,
                  const _Float16* h1_rd, _Float16* h1_wr) {
    if (tid < 64) {
      xs_wr[tid] = pack2f16(rx0, rx1);          // data for step t+1
      int tn = (t + 2) & 2047;                  // wrap: t=2046/2047 stores unused
      int row = tid >> 2, pp = tid & 3, k0 = pp << 1;
      const float* xr = x + xbase + (long)row * (7 * 2048);
      rx0 = xr[k0 * 2048 + tn];
      rx1 = (k0 + 1 < 7) ? xr[(k0 + 1) * 2048 + tn] : 0.f;
    }
    // ---------------- layer 0 ----------------
    half16_t a0lo, a0hi;
    loadA64(h0_rd, laneM, hf, a0lo, a0hi);
    u32x4 qx = ((const u32x4*)xs_rd)[laneM];
    FragU axu;
    axu.u[0] = hf ? 0u : qx.x; axu.u[1] = hf ? 0u : qx.y;
    axu.u[2] = hf ? 0u : qx.z; axu.u[3] = hf ? 0u : qx.w;
    axu.u[4] = 0u; axu.u[5] = 0u; axu.u[6] = 0u; axu.u[7] = 0u;

    float8_t acc[4];
#pragma unroll
    for (int g = 0; g < 4; ++g) {
      float8_t a = wmma16(axu.h, bWx[g], b0v[g]);   // x @ Wih0^T + b
      a = wmma16(a0lo, bW0[g][0], a);               // h0 @ Whh0^T, k=0..31
      a = wmma16(a0hi, bW0[g][1], a);               // k=32..63
      acc[g] = a;
    }
    gates_to_h(acc, c0r, h0_wr, j, hf);

    __syncthreads();   // the ONLY barrier: new h0 (and staged x) visible

    // ---------------- layer 1 ----------------
    half16_t n0lo, n0hi, a1lo, a1hi;
    loadA64(h0_wr, laneM, hf, n0lo, n0hi);   // new h0
    loadA64(h1_rd, laneM, hf, a1lo, a1hi);   // previous h1 (other buffer: no WAR)
#pragma unroll
    for (int g = 0; g < 4; ++g) {
      float8_t a = wmma16(n0lo, bW1i[g][0], b1v[g]);
      a = wmma16(n0hi, bW1i[g][1], a);
      a = wmma16(a1lo, bW1h[g][0], a);
      a = wmma16(a1hi, bW1h[g][1], a);
      acc[g] = a;
    }
    gates_to_h(acc, c1r, h1_wr, j, hf);
  };

  for (int t = 0; t < 2048; t += 2) {
    step(t,     xs0, xs1, h0s0, h0s1, h1s0, h1s1);
    step(t + 1, xs1, xs0, h0s1, h0s0, h1s1, h1s0);
  }

  __syncthreads();   // final h1 lives in h1s0 (written by step 2047)

  // ---------------- fused head: Linear(64->10) + BN + ReLU + Linear(10->1) ----
  if (tid < 16) {
    float o = b2[0];
    for (int jj = 0; jj < 10; ++jj) {
      float s = b1l[jj];
#pragma unroll
      for (int k = 0; k < 64; ++k)
        s += (float)h1s0[(tid << 6) + k] * W1[jj * 64 + k];
      s = (s - rm[jj]) * rsqrtf(rv[jj] + 1e-5f) * gma[jj] + bta[jj];
      s = fmaxf(s, 0.f);
      o += s * W2[jj];
    }
    out[wg * 16 + tid] = o;
  }
}

extern "C" void kernel_launch(void* const* d_in, const int* in_sizes, int n_in,
                              void* d_out, int out_size, void* d_ws, size_t ws_size,
                              hipStream_t stream) {
  (void)in_sizes; (void)n_in; (void)out_size; (void)d_ws; (void)ws_size;
  const float* x    = (const float*)d_in[0];
  const float* Wih0 = (const float*)d_in[1];
  const float* Whh0 = (const float*)d_in[2];
  const float* bih0 = (const float*)d_in[3];
  const float* bhh0 = (const float*)d_in[4];
  const float* Wih1 = (const float*)d_in[5];
  const float* Whh1 = (const float*)d_in[6];
  const float* bih1 = (const float*)d_in[7];
  const float* bhh1 = (const float*)d_in[8];
  const float* W1   = (const float*)d_in[9];
  const float* b1l  = (const float*)d_in[10];
  const float* gma  = (const float*)d_in[11];
  const float* bta  = (const float*)d_in[12];
  const float* rm   = (const float*)d_in[13];
  const float* rv   = (const float*)d_in[14];
  const float* W2   = (const float*)d_in[15];
  const float* b2   = (const float*)d_in[16];

  lstm_fused_kernel<<<dim3(16), dim3(128), 0, stream>>>(
      x, Wih0, Whh0, bih0, bhh0, Wih1, Whh1, bih1, bhh1,
      W1, b1l, gma, bta, rm, rv, W2, b2, (float*)d_out);
}
